// GRUBaseline_25194278158701
// MI455X (gfx1250) — matrix operand order repaired and checked by reading, not compiled, not run on hardware
//
#include <hip/hip_runtime.h>
#include <hip/hip_bf16.h>

typedef __attribute__((ext_vector_type(16))) _Float16 v16h;
typedef __attribute__((ext_vector_type(8)))  _Float16 v8h;
typedef __attribute__((ext_vector_type(8)))  float    v8f;

#define T_STEPS 365
#define NCELLS  (256*256)
#define HID     64
#define GROWS   256   // augmented gate rows: r(64) z(64) in(64) hn(64)
#define KAUG    96    // augmented K: [1, p, tm, h0..h63, zero pad] -> 3 chunks of 32

// ---- cheap transcendentals: gates don't need IEEE-correct division ----
__device__ __forceinline__ float frcp_(float x) { return __builtin_amdgcn_rcpf(x); }

#if __has_builtin(__builtin_amdgcn_tanhf)
__device__ __forceinline__ float ftanh_(float x) { return __builtin_amdgcn_tanhf(x); }
__device__ __forceinline__ float fsig(float x)   { return __builtin_amdgcn_tanhf(0.5f * x) * 0.5f + 0.5f; }
#else
__device__ __forceinline__ float fsig(float x)   { return frcp_(1.0f + __expf(-x)); }
__device__ __forceinline__ float ftanh_(float x) { return 2.0f * frcp_(1.0f + __expf(-2.0f * x)) - 1.0f; }
#endif

__global__ __launch_bounds__(128) void gru_wmma_kernel(
    const float* __restrict__ precip, const float* __restrict__ temp,
    const float* __restrict__ wih,    const float* __restrict__ whh,
    const float* __restrict__ bias,   const float* __restrict__ bias_n,
    const float* __restrict__ outw,   const float* __restrict__ outb,
    const float* __restrict__ inith,  float* __restrict__ out)
{
    __shared__ __align__(64) _Float16 waug[GROWS][KAUG];   // 48 KB, block-shared weights
    __shared__ __align__(64) _Float16 haug[4][16][KAUG];   // 12 KB, per-wave augmented state

    const int tid = threadIdx.x;

    // ---- Stage augmented weight matrix into LDS (f16), once ----
    for (int idx = tid; idx < GROWS * KAUG; idx += blockDim.x) {
        int row = idx / KAUG, col = idx % KAUG;
        float v = 0.0f;
        if (row < 128) {                       // r,z rows: bias + W_ih + W_hh
            if      (col == 0) v = bias[row];
            else if (col == 1) v = wih[row * 2 + 0];
            else if (col == 2) v = wih[row * 2 + 1];
            else if (col < 3 + HID) v = whh[row * HID + (col - 3)];
        } else if (row < 192) {                // in_ rows: bias + W_ih only
            if      (col == 0) v = bias[row];
            else if (col == 1) v = wih[row * 2 + 0];
            else if (col == 2) v = wih[row * 2 + 1];
        } else {                               // hn rows: bias_n + W_hh(n) only
            int k = row - 192;
            if      (col == 0) v = bias_n[k];
            else if (col >= 3 && col < 3 + HID) v = whh[(128 + k) * HID + (col - 3)];
        }
        waug[row][col] = (_Float16)v;
    }
    __syncthreads();

    const int wave = tid >> 5;
    const int lane = tid & 31;
    const int s    = (lane >> 4) & 1;   // half-wave selector
    const int cr   = lane & 15;         // cell-in-wave == N index == A-row index
    const int cell = blockIdx.x * 64 + wave * 16 + cr;

    _Float16* hrow = &haug[wave][cr][0];

    // ---- Init per-cell augmented state vector ----
    if (s == 0) {
        hrow[0] = (_Float16)1.0f;                                       // bias lane
        for (int i = 3 + HID; i < KAUG; ++i) hrow[i] = (_Float16)0.0f;  // zero pad
    }
    float hreg[32];   // f32 master copy of hidden state (this lane's 32 k's)
    float owreg[32];  // out_w values for this lane's 32 k's (kept in VGPRs)
    const float ob = outb[0];
#pragma unroll
    for (int tt = 0; tt < 4; ++tt)
#pragma unroll
        for (int j = 0; j < 8; ++j) {
            int k = 16 * tt + j + 8 * s;
            float h0 = inith[k];
            hreg[tt * 8 + j]  = h0;
            owreg[tt * 8 + j] = outw[k];
            hrow[3 + k] = (_Float16)h0;
        }

    const v8f vzero = {0.f, 0.f, 0.f, 0.f, 0.f, 0.f, 0.f, 0.f};
    float* finalH = out + (size_t)T_STEPS * NCELLS;

    // ================= recurrent time loop =================
    for (int t = 0; t < T_STEPS; ++t) {
        float p  = precip[(size_t)t * NCELLS + cell];
        float tm = temp  [(size_t)t * NCELLS + cell];
        if (s == 0) { hrow[1] = (_Float16)p; hrow[2] = (_Float16)tm; }

        int tn = (t + 2 < T_STEPS) ? (t + 2) : t;   // prefetch 2 steps ahead
        __builtin_prefetch(&precip[(size_t)tn * NCELLS + cell], 0, 3);
        __builtin_prefetch(&temp  [(size_t)tn * NCELLS + cell], 0, 3);

        // B fragments: haug columns for this lane's cell; 32x16 f16 layout:
        // lanes 0-15 hold K{0..7,16..23} (N=lane), lanes 16-31 hold K{8..15,24..31}.
        v16h bf[3];
#pragma unroll
        for (int c = 0; c < 3; ++c) {
            const _Float16* bp = &haug[wave][cr][32 * c + 8 * s];
            v8h blo = *(const v8h*)bp;
            v8h bhi = *(const v8h*)(bp + 16);
            bf[c] = __builtin_shufflevector(blo, bhi, 0,1,2,3,4,5,6,7,8,9,10,11,12,13,14,15);
        }

        v8f acc[16];
#pragma unroll
        for (int m = 0; m < 16; ++m) acc[m] = vzero;

        // D(16 gates x 16 cells) += A(16x32 weights) * B(32x16 haug)
        // A fragments are loop-invariant -> compiler hoists them into VGPRs.
#pragma unroll
        for (int mt = 0; mt < 16; ++mt) {
#pragma unroll
            for (int c = 0; c < 3; ++c) {
                if (mt >= 8 && mt < 12 && c > 0) continue;  // in_ rows: chunk 0 only
                // A 16x32 f16 layout: lanes<16 hold K{0..7,16..23}, lanes>=16 K{8..15,24..31}
                const _Float16* ap = &waug[16 * mt + cr][32 * c + 8 * s];
                v8h alo = *(const v8h*)ap;
                v8h ahi = *(const v8h*)(ap + 16);
                v16h a = __builtin_shufflevector(alo, ahi,
                          0,1,2,3,4,5,6,7,8,9,10,11,12,13,14,15);
                acc[mt] = __builtin_amdgcn_wmma_f32_16x16x32_f16(
                              false, a, false, bf[c], (short)0, acc[mt], false, false);
            }
        }

        // ---- element-wise GRU update; D layout: gate = 16*mt + j + 8*s, cell = cr ----
        float yacc = 0.0f;
#pragma unroll
        for (int tt = 0; tt < 4; ++tt)
#pragma unroll
            for (int j = 0; j < 8; ++j) {
                float r  = fsig(acc[tt    ][j]);       // sigmoid(ir+hr+b)
                float z  = fsig(acc[4 + tt][j]);       // sigmoid(iz+hz+b)
                float nn = ftanh_(acc[8 + tt][j] + r * acc[12 + tt][j]); // tanh(in_ + r*(hn+bn))
                float hp = hreg[tt * 8 + j];
                float hx = nn + z * (hp - nn);
                hreg[tt * 8 + j] = hx;
                int k = 16 * tt + j + 8 * s;
                hrow[3 + k] = (_Float16)hx;            // feed next step's B
                yacc = fmaf(owreg[tt * 8 + j], hx, yacc);
            }
        yacc += __shfl_xor(yacc, 16, 32);              // combine lane pair (same cell)
        if (s == 0) out[(size_t)t * NCELLS + cell] = yacc + ob;
    }

    // ---- final hidden state: (H, W, HID) = cell-major, k-minor ----
#pragma unroll
    for (int tt = 0; tt < 4; ++tt)
#pragma unroll
        for (int j = 0; j < 8; ++j) {
            int k = 16 * tt + j + 8 * s;
            finalH[(size_t)cell * HID + k] = hreg[tt * 8 + j];
        }
}

extern "C" void kernel_launch(void* const* d_in, const int* in_sizes, int n_in,
                              void* d_out, int out_size, void* d_ws, size_t ws_size,
                              hipStream_t stream) {
    gru_wmma_kernel<<<NCELLS / 64, 128, 0, stream>>>(
        (const float*)d_in[0],  // precipitation (T,H,W)
        (const float*)d_in[1],  // temperature   (T,H,W)
        (const float*)d_in[2],  // weight_ih (192,2)
        (const float*)d_in[3],  // weight_hh (192,64)
        (const float*)d_in[4],  // bias (192)
        (const float*)d_in[5],  // bias_n (64)
        (const float*)d_in[6],  // out_w (1,64)
        (const float*)d_in[7],  // out_b (1)
        (const float*)d_in[8],  // init_h (64)
        (float*)d_out);         // smb (T,H,W) ++ final_h (H,W,64)
}
